// AMPBlock0_33517924778396
// MI455X (gfx1250) — matrix operand order repaired and checked by reading, not compiled, not run on hardware
//
#include <hip/hip_runtime.h>

// AMPBlock0 forward for MI455X (gfx1250, wave32, WMMA).
// Pipeline: act_snake -> conv3x1_wmma(bf16) -> act_snake -> conv3x1_wmma(+residual)

#define B_ 8
#define C_ 512
#define T_ 8192

typedef __attribute__((ext_vector_type(16))) __bf16 v16bf;
typedef __attribute__((ext_vector_type(8)))  float  v8f;

struct __align__(16) U4 { unsigned int x, y, z, w; };
struct FragBytes { U4 lo, hi; };   // 32 bytes == one 16-element bf16 WMMA fragment

__device__ __forceinline__ unsigned short f2bf(float f) {
  unsigned int u = __float_as_uint(f);
  u += 0x7FFFu + ((u >> 16) & 1u);          // round-to-nearest-even
  return (unsigned short)(u >> 16);
}

// ---------------------------------------------------------------------------
// Weight prep: (C,C,3) f32 -> (3,C,C) bf16 so K (=ci) is contiguous per tap.
// ---------------------------------------------------------------------------
__global__ __launch_bounds__(256) void prep_w(const float* __restrict__ w,
                                              unsigned short* __restrict__ wt) {
  int o = blockIdx.x * 256 + threadIdx.x;
  if (o < 3 * C_ * C_) {
    int d  = o / (C_ * C_);
    int r  = o - d * C_ * C_;
    int co = r >> 9;
    int ci = r & (C_ - 1);
    wt[o] = f2bf(w[(co * C_ + ci) * 3 + d]);
  }
}

// ---------------------------------------------------------------------------
// Fused Activation1d: polyphase x2 upsample (k=6) -> snake-beta -> stride-2
// lowpass (k=12, pad=5).  In: (B,C,Tin) f32, t-contiguous.
// Out: (B,Tout,C) bf16 (transposed layout for the WMMA GEMM B-tiles).
// z[t_o] = sum_j dw[j] * V(2*t_o - 5 + j),  V(s)=snake(u(s)),
// u(s)   = sum_k x[floor(s/2) - 1 + k] * up_w[2c + (s&1)][k], s in [0, 2Tin-5].
// ---------------------------------------------------------------------------
__global__ __launch_bounds__(256) void act_snake(
    const float* __restrict__ x,       // (B,C,Tin)
    const float* __restrict__ upw,     // (2C,6)
    const float* __restrict__ dnw,     // (C,12)
    const float* __restrict__ alpha,   // (C)
    const float* __restrict__ beta,    // (C)
    unsigned short* __restrict__ zout, // (B,Tout,C) bf16
    int Tin, int Tout)
{
  __shared__ float          xs[32][144];  // x tile [c][t0-4 .. t0+134]
  __shared__ unsigned short zb[128][40];  // transpose buffer, 80B rows (16B aligned)

  const int b  = blockIdx.z;
  const int c0 = blockIdx.y * 32;
  const int t0 = blockIdx.x * 128;

  for (int idx = threadIdx.x; idx < 32 * 144; idx += 256) {
    int cl = idx / 144, tl = idx - cl * 144;
    int g  = t0 - 4 + tl;
    float v = 0.f;
    if (g >= 0 && g < Tin) v = x[((size_t)(b * C_ + c0 + cl)) * Tin + g];
    xs[cl][tl] = v;
  }
  __syncthreads();

  const int cl = threadIdx.x >> 3;        // channel within tile (0..31)
  const int tb = (threadIdx.x & 7) * 16;  // 16 outputs per thread
  const int c  = c0 + cl;

  float uw0[6], uw1[6], dw[12];
#pragma unroll
  for (int j = 0; j < 6; ++j)  { uw0[j] = upw[(2 * c) * 6 + j]; uw1[j] = upw[(2 * c + 1) * 6 + j]; }
#pragma unroll
  for (int j = 0; j < 12; ++j) dw[j] = dnw[c * 12 + j];
  const float aa    = __expf(alpha[c]);
  const float inv2b = 1.f / (2.f * __expf(beta[c]) + 1e-9f);

  const int tbase = t0 + tb;
  const int smax  = 2 * Tin - 5;

  float V[42];                             // upsampled+snaked values, s = 2*tbase-5+k
#pragma unroll
  for (int k = 0; k < 42; ++k) {
    const int s = 2 * tbase - 5 + k;       // parity of s == parity of (k+1): folds
    float val = 0.f;
    if (s >= 0 && s <= smax) {
      const int q = s >> 1;
      const float* xp = &xs[cl][q + 3 - t0];
      const float* w  = (s & 1) ? uw1 : uw0;
      float u = 0.f;
#pragma unroll
      for (int j = 0; j < 6; ++j) u = fmaf(xp[j], w[j], u);
      val = u + (1.f - __cosf(2.f * aa * u)) * inv2b;
    }
    V[k] = val;
  }

#pragma unroll
  for (int i = 0; i < 16; ++i) {
    float zv = 0.f;
#pragma unroll
    for (int j = 0; j < 12; ++j) zv = fmaf(dw[j], V[2 * i + j], zv);
    zb[tb + i][cl] = f2bf(zv);
  }
  __syncthreads();

  // coalesced transposed store: each row t is 32 consecutive channels (64B)
  for (int idx = threadIdx.x; idx < 128 * 4; idx += 256) {
    int row = idx >> 2, ch = idx & 3;
    int t = t0 + row;
    if (t < Tout) {
      *(U4*)((char*)zout + (((size_t)(b * Tout + t)) * C_ + c0) * 2 + ch * 16) =
          *(const U4*)(&zb[row][ch * 8]);
    }
  }
}

// ---------------------------------------------------------------------------
// Dense conv (C->C, k=3, pad=1) as 3 accumulated bf16 GEMMs via WMMA.
// out[b,co,t] = bias[co] + sum_d sum_ci Wt[d][co][ci] * z[b, t+d-1, ci] (+res)
// Workgroup: 64 Cout x 64 t, 8 waves; wave (mw,nw) does M-tile mw, N-tiles
// {2nw, 2nw+1}.  Z tile staged in LDS in two K-halves (rows padded +16B).
// ---------------------------------------------------------------------------
template<bool RES>
__global__ __launch_bounds__(256) void conv3x1_wmma(
    const unsigned short* __restrict__ z,    // (B,Tz,C) bf16
    const unsigned short* __restrict__ wt,   // (3,C,C) bf16
    const float* __restrict__ bias,          // (C)
    const float* __restrict__ resid,         // (B,C,Tres) f32
    float* __restrict__ out,                 // (B,C,Tout) f32
    int Tz, int Tout, int Tres)
{
  __shared__ unsigned short zt[66 * 264];    // 66 t-rows x 256 ci (+8 pad) bf16

  const int b    = blockIdx.z;
  const int t0   = blockIdx.x * 64;
  const int coB  = blockIdx.y * 64;

  const int lane = threadIdx.x & 31;
  const int wv   = threadIdx.x >> 5;
  const int mw   = wv & 3;                   // M-tile (16 Cout rows)
  const int nw   = wv >> 2;                  // N half (2 x 16 t cols)
  const int hi   = lane >> 4;                // lane half selects K group
  const int lm   = lane & 15;

  v8f acc0 = {}; v8f acc1 = {};

  for (int kh = 0; kh < 2; ++kh) {           // K (=ci) halves of 256
    __syncthreads();
    for (int idx = threadIdx.x; idx < 66 * 32; idx += 256) {
      int r = idx >> 5, ch = idx & 31;
      int t = t0 - 1 + r;                    // rows cover [t0-1, t0+64] (pad=1 halo)
      U4 val = {0u, 0u, 0u, 0u};
      if (t >= 0 && t < Tz)
        val = *(const U4*)(z + ((size_t)(b * Tz + t) * C_ + kh * 256 + ch * 8));
      *(U4*)(&zt[r * 264 + ch * 8]) = val;   // zero-pad conv boundaries
    }
    __syncthreads();

    for (int d = 0; d < 3; ++d) {
      // A: weights for rows coB+mw*16+lm, tap d (contiguous K, L2-resident)
      const unsigned short* wr =
          wt + ((size_t)d * C_ + coB + mw * 16 + lm) * C_ + kh * 256;
      __builtin_prefetch(wr, 0, 1);          // global_prefetch_b8
      // B: column t = t0 + c_loc + (d-1)  ->  LDS row r = c_loc + d
      const unsigned short* zr0 = &zt[((nw * 2 + 0) * 16 + lm + d) * 264 + hi * 16];
      const unsigned short* zr1 = &zt[((nw * 2 + 1) * 16 + lm + d) * 264 + hi * 16];

      for (int kc = 0; kc < 8; ++kc) {
        const int kb = kc * 32;
        FragBytes fa, f0, f1;
        // A layout matching B: lanes 0-15 hold K{0-15}, lanes 16-31 K{16-31}
        fa.lo = *(const U4*)(wr + kb + hi * 16);
        fa.hi = *(const U4*)(wr + kb + hi * 16 + 8);
        // B layout (same K order as A): lane = N col, lane half = K group of 16 (contiguous)
        f0.lo = *(const U4*)(zr0 + kb);
        f0.hi = *(const U4*)(zr0 + kb + 8);
        f1.lo = *(const U4*)(zr1 + kb);
        f1.hi = *(const U4*)(zr1 + kb + 8);
        v16bf a  = __builtin_bit_cast(v16bf, fa);
        v16bf b0 = __builtin_bit_cast(v16bf, f0);
        v16bf b1 = __builtin_bit_cast(v16bf, f1);
        acc0 = __builtin_amdgcn_wmma_f32_16x16x32_bf16(false, a, false, b0,
                                                       (short)0, acc0, false, false);
        acc1 = __builtin_amdgcn_wmma_f32_16x16x32_bf16(false, a, false, b1,
                                                       (short)0, acc1, false, false);
      }
    }
  }

  // Epilogue.  C/D layout: VGPR v, lanes 0-15 -> M=v, N=lane; lanes 16-31 -> M=v+8.
  for (int j = 0; j < 2; ++j) {
    const v8f acc = j ? acc1 : acc0;
    const int t = t0 + (nw * 2 + j) * 16 + lm;
    if (t < Tout) {
#pragma unroll
      for (int v = 0; v < 8; ++v) {
        const int co = coB + mw * 16 + v + hi * 8;
        float val = acc[v] + bias[co];
        if (RES) val += resid[((size_t)b * C_ + co) * Tres + t];
        out[((size_t)b * C_ + co) * Tout + t] = val;
      }
    }
  }
}

// ---------------------------------------------------------------------------
extern "C" void kernel_launch(void* const* d_in, const int* in_sizes, int n_in,
                              void* d_out, int out_size, void* d_ws, size_t ws_size,
                              hipStream_t stream) {
  (void)in_sizes; (void)n_in; (void)out_size; (void)ws_size;

  const float* x   = (const float*)d_in[0];
  const float* up1 = (const float*)d_in[1];
  const float* dn1 = (const float*)d_in[2];
  const float* al1 = (const float*)d_in[3];
  const float* be1 = (const float*)d_in[4];
  const float* up2 = (const float*)d_in[5];
  const float* dn2 = (const float*)d_in[6];
  const float* al2 = (const float*)d_in[7];
  const float* be2 = (const float*)d_in[8];
  const float* c1w = (const float*)d_in[9];
  const float* c1b = (const float*)d_in[10];
  const float* c2w = (const float*)d_in[11];
  const float* c2b = (const float*)d_in[12];

  const int T1 = T_ - 2;   // 8190
  const int T2 = T_ - 4;   // 8188

  char* ws = (char*)d_ws;
  size_t off = 0;
  auto take = [&](size_t bytes) {
    char* p = ws + off;
    off += (bytes + 255) & ~(size_t)255;
    return p;
  };
  unsigned short* w1t = (unsigned short*)take((size_t)3 * C_ * C_ * 2);
  unsigned short* w2t = (unsigned short*)take((size_t)3 * C_ * C_ * 2);
  unsigned short* z1  = (unsigned short*)take((size_t)B_ * T1 * C_ * 2);
  float*          y1  = (float*)take((size_t)B_ * C_ * T1 * 4);
  unsigned short* z2  = (unsigned short*)take((size_t)B_ * T2 * C_ * 2);

  prep_w<<<3 * C_ * C_ / 256, 256, 0, stream>>>(c1w, w1t);
  prep_w<<<3 * C_ * C_ / 256, 256, 0, stream>>>(c2w, w2t);

  dim3 gact(64, C_ / 32, B_);    // 64 t-blocks of 128 cover 8190/8188
  dim3 gcnv(128, C_ / 64, B_);   // 128 t-blocks of 64

  act_snake<<<gact, 256, 0, stream>>>(x,  up1, dn1, al1, be1, z1, T_, T1);
  conv3x1_wmma<false><<<gcnv, 256, 0, stream>>>(z1, w1t, c1b, nullptr, y1, T1, T1, 0);
  act_snake<<<gact, 256, 0, stream>>>(y1, up2, dn2, al2, be2, z2, T1, T2);
  conv3x1_wmma<true ><<<gcnv, 256, 0, stream>>>(z2, w2t, c2b, x, (float*)d_out, T2, T2, T_);
}
